// MultiHeadAttention_49950469653274
// MI455X (gfx1250) — matrix operand order repaired and checked by reading, not compiled, not run on hardware
//
#include <hip/hip_runtime.h>
#include <hip/hip_bf16.h>

// CDNA5 (gfx1250) wave32 WMMA types
typedef __bf16 v16bf __attribute__((ext_vector_type(16)));
typedef float  v8f   __attribute__((ext_vector_type(8)));

#define TK 32
#define LDPAD 8   // LDS row padding (bf16 elems); keeps 16B alignment (80B rows)

// ---------------------------------------------------------------------------
// Fragment loaders from LDS tiles, in contiguous k order (k = 16*khalf + s), same for A and B.
// A (16x32, MxK): lane = m + 16*khalf; VGPR j holds K=16*kh+2j (+1).
// B (32x16, KxN): lane = n + 16*khalf; VGPR j holds K=16*kh+2j (+1).
// Tiles are stored row-major [row][k]; loads merge into ds_load_b128.
// ---------------------------------------------------------------------------
__device__ __forceinline__ v16bf frag_a(const __bf16* t, int row0, int ldt, int lane) {
  const int m = lane & 15, kh = lane >> 4;
  const __bf16* r = t + (row0 + m) * ldt + 16 * kh;
  v16bf a;
#pragma unroll
  for (int j = 0; j < 8; ++j) {
    const int kb = 2 * j;
    a[2 * j]     = r[kb];
    a[2 * j + 1] = r[kb + 1];
  }
  return a;
}

__device__ __forceinline__ v16bf frag_b(const __bf16* t, int n0, int ldt, int lane) {
  const int n = lane & 15, kh = lane >> 4;
  const __bf16* r = t + (n0 + n) * ldt + 16 * kh;
  v16bf b;
#pragma unroll
  for (int j = 0; j < 8; ++j) {
    b[2 * j]     = r[2 * j];
    b[2 * j + 1] = r[2 * j + 1];
  }
  return b;
}

// ---------------------------------------------------------------------------
// C = scale * (A @ B^T) + bias     (A: MxK bf16, B: NxK bf16, row-major)
// Workgroup tile 128x128, 8 waves (4Mx2N) of 32x64 each (2x4 accumulators).
// 128-bit global->reg->LDS fills, register double-buffered K pipeline.
// Batched over blockIdx.z: offset = (z/zmod)*s?b + (z%zmod)*s?h for A/B,
// linear z*sCz for C.
// ---------------------------------------------------------------------------
template <typename OT>
__global__ __launch_bounds__(256) void gemm_nt(
    const __bf16* __restrict__ A, long long lda, long long sAb, long long sAh,
    const __bf16* __restrict__ Bm, long long ldb, long long sBb, long long sBh,
    OT* __restrict__ C, long long ldc, long long sCz,
    const float* __restrict__ bias, int K, float scale, int zmod)
{
  __shared__ __bf16 As[128][TK + LDPAD];
  __shared__ __bf16 Bs[128][TK + LDPAD];
  const int tid = threadIdx.x, lane = tid & 31, wave = tid >> 5;
  const int wm = (wave >> 1) * 32, wn = (wave & 1) * 64;
  const int z = blockIdx.z, zb = z / zmod, zh = z % zmod;

  const __bf16* Ag = A + zb * sAb + zh * sAh + (long long)blockIdx.x * 128 * lda;
  const __bf16* Bg = Bm + zb * sBb + zh * sBh + (long long)blockIdx.y * 128 * ldb;
  OT* Cg = C + (long long)z * sCz + (long long)blockIdx.x * 128 * ldc
             + (long long)blockIdx.y * 128;

  // Per-thread 16B chunks: row r0 (+64), 8 bf16 starting at kq.
  const int r0 = tid >> 2;        // 0..63
  const int kq = (tid & 3) * 8;   // 0,8,16,24

  v8f acc[2][4];
#pragma unroll
  for (int i = 0; i < 2; ++i)
#pragma unroll
    for (int j = 0; j < 4; ++j)
#pragma unroll
      for (int e = 0; e < 8; ++e) acc[i][j][e] = 0.0f;

  // prologue: first K-slab into registers
  uint4 a0 = *(const uint4*)&Ag[(long long)r0 * lda + kq];
  uint4 a1 = *(const uint4*)&Ag[(long long)(r0 + 64) * lda + kq];
  uint4 b0 = *(const uint4*)&Bg[(long long)r0 * ldb + kq];
  uint4 b1 = *(const uint4*)&Bg[(long long)(r0 + 64) * ldb + kq];

  for (int k0 = 0; k0 < K; k0 += TK) {
    *(uint4*)&As[r0][kq]      = a0;   // ds_store_b128
    *(uint4*)&As[r0 + 64][kq] = a1;
    *(uint4*)&Bs[r0][kq]      = b0;
    *(uint4*)&Bs[r0 + 64][kq] = b1;
    __syncthreads();

    const bool more = (k0 + TK) < K;
    uint4 na0{}, na1{}, nb0{}, nb1{};
    if (more) {  // next slab's global loads in flight during WMMA
      const long long kn = (long long)(k0 + TK) + kq;
      na0 = *(const uint4*)&Ag[(long long)r0 * lda + kn];
      na1 = *(const uint4*)&Ag[(long long)(r0 + 64) * lda + kn];
      nb0 = *(const uint4*)&Bg[(long long)r0 * ldb + kn];
      nb1 = *(const uint4*)&Bg[(long long)(r0 + 64) * ldb + kn];
    }

    v16bf af[2], bfv[4];
#pragma unroll
    for (int i = 0; i < 2; ++i) af[i]  = frag_a(&As[0][0], wm + 16 * i, TK + LDPAD, lane);
#pragma unroll
    for (int j = 0; j < 4; ++j) bfv[j] = frag_b(&Bs[0][0], wn + 16 * j, TK + LDPAD, lane);
#pragma unroll
    for (int i = 0; i < 2; ++i)
#pragma unroll
      for (int j = 0; j < 4; ++j)
        acc[i][j] = __builtin_amdgcn_wmma_f32_16x16x32_bf16(
            false, af[i], false, bfv[j], (short)0, acc[i][j], false, false);
    __syncthreads();

    if (more) { a0 = na0; a1 = na1; b0 = nb0; b1 = nb1; }
  }

  // C/D layout: lane = n + 16*mhalf, VGPR r -> m = r + 8*mhalf
  const int cn = lane & 15, ch = lane >> 4;
#pragma unroll
  for (int i = 0; i < 2; ++i)
#pragma unroll
    for (int j = 0; j < 4; ++j) {
      const int gn = wn + 16 * j + cn;
      const float bvv = bias ? bias[(long long)blockIdx.y * 128 + gn] : 0.0f;
#pragma unroll
      for (int r = 0; r < 8; ++r) {
        const int gm = wm + 16 * i + 8 * ch + r;
        Cg[(long long)gm * ldc + gn] = (OT)(acc[i][j][r] * scale + bvv);
      }
    }
}

// ---------------------------------------------------------------------------
// C = A @ B  (A: MxK f32 -> bf16 in-flight, B: KxN bf16 row-major, N tile=64)
// Workgroup tile 256x64, 8 waves (4Mx2N) of 64x32 each (4x2 accumulators).
// A: float4 loads -> packed bf16 uint2 LDS stores. B: uint4 row loads,
// scatter-transposed into [n][k] so both fragments use the fast layout.
// Used for ctx = P @ V per (b,h); P read from d_out attn region.
// ---------------------------------------------------------------------------
__global__ __launch_bounds__(256) void gemm_nn(
    const float* __restrict__ A, long long lda, long long sAz,
    const __bf16* __restrict__ Bm, long long ldb, long long sBb, long long sBh,
    __bf16* __restrict__ C, long long ldc, long long sCb, long long sCh,
    int K, int zmod)
{
  __shared__ __bf16 As[256][TK + LDPAD];   // 20 KB
  __shared__ __bf16 Bs[64][TK + LDPAD];    //  5 KB  (stored [n][k])
  const int tid = threadIdx.x, lane = tid & 31, wave = tid >> 5;
  const int wm = (wave >> 1) * 64, wn = (wave & 1) * 32;
  const int z = blockIdx.z, zb = z / zmod, zh = z % zmod;

  const float*  Ag = A + (long long)z * sAz + (long long)blockIdx.x * 256 * lda;
  const __bf16* Bg = Bm + zb * sBb + zh * sBh;
  __bf16*       Cg = C + zb * sCb + zh * sCh + (long long)blockIdx.x * 256 * ldc;

  // A: 256x32 f32 = 8 float4/thread: rows ra+32*l, k quad ka.
  const int ra = tid >> 3;        // 0..31
  const int ka = (tid & 7) * 4;   // 0,4,..,28
  // B: 32x64 bf16 = 1 uint4/thread: k-row kb, 8 cols from nb0.
  const int kb = tid >> 3;        // 0..31
  const int nb0 = (tid & 7) * 8;  // 0,8,..,56

  v8f acc[4][2];
#pragma unroll
  for (int i = 0; i < 4; ++i)
#pragma unroll
    for (int j = 0; j < 2; ++j)
#pragma unroll
      for (int e = 0; e < 8; ++e) acc[i][j][e] = 0.0f;

  float4 av[8];
#pragma unroll
  for (int l = 0; l < 8; ++l)
    av[l] = *(const float4*)&Ag[(long long)(ra + 32 * l) * lda + ka];
  uint4 bvu = *(const uint4*)&Bg[(long long)kb * ldb + nb0];

  for (int k0 = 0; k0 < K; k0 += TK) {
#pragma unroll
    for (int l = 0; l < 8; ++l) {   // convert + packed 8B LDS stores
      union { uint2 u; __bf16 h[4]; } pk;
      pk.h[0] = (__bf16)av[l].x; pk.h[1] = (__bf16)av[l].y;
      pk.h[2] = (__bf16)av[l].z; pk.h[3] = (__bf16)av[l].w;
      *(uint2*)&As[ra + 32 * l][ka] = pk.u;
    }
    {
      union { uint4 u; __bf16 h[8]; } pb;
      pb.u = bvu;
#pragma unroll
      for (int j = 0; j < 8; ++j) Bs[nb0 + j][kb] = pb.h[j];
    }
    __syncthreads();

    const bool more = (k0 + TK) < K;
    float4 nav[8]; uint4 nbv{};
#pragma unroll
    for (int l = 0; l < 8; ++l) nav[l] = make_float4(0.f, 0.f, 0.f, 0.f);
    if (more) {
      const long long kn = (long long)(k0 + TK);
#pragma unroll
      for (int l = 0; l < 8; ++l)
        nav[l] = *(const float4*)&Ag[(long long)(ra + 32 * l) * lda + kn + ka];
      nbv = *(const uint4*)&Bg[(kn + kb) * ldb + nb0];
    }

    v16bf af[4], bfv[2];
#pragma unroll
    for (int i = 0; i < 4; ++i) af[i]  = frag_a(&As[0][0], wm + 16 * i, TK + LDPAD, lane);
#pragma unroll
    for (int j = 0; j < 2; ++j) bfv[j] = frag_b(&Bs[0][0], wn + 16 * j, TK + LDPAD, lane);
#pragma unroll
    for (int i = 0; i < 4; ++i)
#pragma unroll
      for (int j = 0; j < 2; ++j)
        acc[i][j] = __builtin_amdgcn_wmma_f32_16x16x32_bf16(
            false, af[i], false, bfv[j], (short)0, acc[i][j], false, false);
    __syncthreads();

    if (more) {
#pragma unroll
      for (int l = 0; l < 8; ++l) av[l] = nav[l];
      bvu = nbv;
    }
  }

  const int cn = lane & 15, ch = lane >> 4;
#pragma unroll
  for (int i = 0; i < 4; ++i)
#pragma unroll
    for (int j = 0; j < 2; ++j) {
      const int gn = wn + 16 * j + cn;
#pragma unroll
      for (int r = 0; r < 8; ++r) {
        const int gm = wm + 16 * i + 8 * ch + r;
        Cg[(long long)gm * ldc + gn] = (__bf16)acc[i][j][r];
      }
    }
}

// ---------------------------------------------------------------------------
// In-place row softmax over rows of length 4096 (one workgroup per row),
// float4 vectorized: 4x b128 loads/stores per thread per pass.
// ---------------------------------------------------------------------------
__global__ __launch_bounds__(256) void softmax4096(float* __restrict__ attn) {
  float4* p4 = (float4*)(attn + (long long)blockIdx.x * 4096);
  __shared__ float red[256];
  const int tid = threadIdx.x;
  float4 v[4];
  float m = -3.402823466e38f;
#pragma unroll
  for (int i = 0; i < 4; ++i) {
    v[i] = p4[tid + 256 * i];
    m = fmaxf(m, fmaxf(fmaxf(v[i].x, v[i].y), fmaxf(v[i].z, v[i].w)));
  }
  red[tid] = m; __syncthreads();
  for (int s = 128; s > 0; s >>= 1) {
    if (tid < s) red[tid] = fmaxf(red[tid], red[tid + s]);
    __syncthreads();
  }
  m = red[0]; __syncthreads();
  float sum = 0.0f;
#pragma unroll
  for (int i = 0; i < 4; ++i) {
    v[i].x = __expf(v[i].x - m); v[i].y = __expf(v[i].y - m);
    v[i].z = __expf(v[i].z - m); v[i].w = __expf(v[i].w - m);
    sum += v[i].x + v[i].y + v[i].z + v[i].w;
  }
  red[tid] = sum; __syncthreads();
  for (int s = 128; s > 0; s >>= 1) {
    if (tid < s) red[tid] += red[tid + s];
    __syncthreads();
  }
  const float inv = 1.0f / red[0];
#pragma unroll
  for (int i = 0; i < 4; ++i) {
    v[i].x *= inv; v[i].y *= inv; v[i].z *= inv; v[i].w *= inv;
    p4[tid + 256 * i] = v[i];
  }
}

// ---------------------------------------------------------------------------
// f32 -> bf16, 8 elems/thread (2x float4 in, 1x uint4 out). n8 = n/8.
// ---------------------------------------------------------------------------
__global__ __launch_bounds__(256) void cvt_bf16(const float* __restrict__ s,
                                                __bf16* __restrict__ d, int n8) {
  const int i = blockIdx.x * 256 + threadIdx.x;
  if (i < n8) {
    const float4* s4 = (const float4*)s;
    const float4 x0 = s4[2 * i], x1 = s4[2 * i + 1];
    union { uint4 u; __bf16 h[8]; } pk;
    pk.h[0] = (__bf16)x0.x; pk.h[1] = (__bf16)x0.y;
    pk.h[2] = (__bf16)x0.z; pk.h[3] = (__bf16)x0.w;
    pk.h[4] = (__bf16)x1.x; pk.h[5] = (__bf16)x1.y;
    pk.h[6] = (__bf16)x1.z; pk.h[7] = (__bf16)x1.w;
    ((uint4*)d)[i] = pk.u;
  }
}

// ---------------------------------------------------------------------------
extern "C" void kernel_launch(void* const* d_in, const int* in_sizes, int n_in,
                              void* d_out, int out_size, void* d_ws, size_t ws_size,
                              hipStream_t stream) {
  (void)in_sizes; (void)n_in; (void)out_size; (void)ws_size;
  constexpr long long Bn = 2, Nn = 4096, Dn = 512, Hn = 8, Dh = 64;
  constexpr long long ACT = Bn * Nn * Dn;   // 4,194,304 activation elems
  constexpr long long WSZ = Dn * Dn;        //   262,144 weight elems

  const float* q_in = (const float*)d_in[0];
  const float* k_in = (const float*)d_in[1];
  const float* v_in = (const float*)d_in[2];
  const float* Wq = (const float*)d_in[3];
  const float* bq = (const float*)d_in[4];
  const float* Wk = (const float*)d_in[5];
  const float* bk = (const float*)d_in[6];
  const float* Wv = (const float*)d_in[7];
  const float* bv = (const float*)d_in[8];
  const float* Wo = (const float*)d_in[9];
  const float* bo = (const float*)d_in[10];

  // Workspace layout (bf16): x{q,k,v}, W{q,k,v,o}, y{q,k,v}, ctx  (~61 MB)
  __bf16* ws  = (__bf16*)d_ws;
  __bf16* xq  = ws;
  __bf16* xk  = xq + ACT;
  __bf16* xv  = xk + ACT;
  __bf16* wqb = xv + ACT;
  __bf16* wkb = wqb + WSZ;
  __bf16* wvb = wkb + WSZ;
  __bf16* wob = wvb + WSZ;
  __bf16* yq  = wob + WSZ;
  __bf16* yk  = yq + ACT;
  __bf16* yv  = yk + ACT;
  __bf16* ctx = yv + ACT;

  float* outp = (float*)d_out;
  float* attn = outp + ACT;   // [B*H, N, N] f32

  auto cv = [&](const float* s, __bf16* d, long long n) {
    const long long n8 = n / 8;
    cvt_bf16<<<dim3((unsigned)((n8 + 255) / 256)), dim3(256), 0, stream>>>(s, d, (int)n8);
  };
  cv(q_in, xq, ACT); cv(k_in, xk, ACT); cv(v_in, xv, ACT);
  cv(Wq, wqb, WSZ);  cv(Wk, wkb, WSZ); cv(Wv, wvb, WSZ); cv(Wo, wob, WSZ);

  // Q/K/V projections: y = x @ W^T + b   [8192x512] x [512x512]
  dim3 gproj(8192 / 128, 512 / 128, 1);
  gemm_nt<__bf16><<<gproj, 256, 0, stream>>>(xq, Dn, 0, 0, wqb, Dn, 0, 0,
                                             yq, Dn, 0, bq, (int)Dn, 1.0f, 1);
  gemm_nt<__bf16><<<gproj, 256, 0, stream>>>(xk, Dn, 0, 0, wkb, Dn, 0, 0,
                                             yk, Dn, 0, bk, (int)Dn, 1.0f, 1);
  gemm_nt<__bf16><<<gproj, 256, 0, stream>>>(xv, Dn, 0, 0, wvb, Dn, 0, 0,
                                             yv, Dn, 0, bv, (int)Dn, 1.0f, 1);

  // scores: S = (1/sqrt(64)) * q_bh @ k_bh^T  -> attn (f32), batched over 16 (b,h)
  dim3 gsc(4096 / 128, 4096 / 128, (unsigned)(Bn * Hn));
  gemm_nt<float><<<gsc, 256, 0, stream>>>(yq, Dn, ACT, Dh, yk, Dn, ACT, Dh,
                                          attn, Nn, Nn * Nn, nullptr,
                                          (int)Dh, 0.125f, (int)Hn);

  // row softmax in place on attn
  softmax4096<<<dim3((unsigned)(Bn * Hn * Nn)), 256, 0, stream>>>(attn);

  // ctx = P @ V per (b,h), written into head slots of [B*N, 512] (bf16)
  dim3 gctx(4096 / 256, 1, (unsigned)(Bn * Hn));
  gemm_nn<<<gctx, 256, 0, stream>>>(attn, Nn, Nn * Nn, yv, Dn, ACT, Dh,
                                    ctx, Dn, ACT, Dh, (int)Nn, (int)Hn);

  // out = ctx @ Wo^T + bo  (f32)
  gemm_nt<float><<<gproj, 256, 0, stream>>>(ctx, Dn, 0, 0, wob, Dn, 0, 0,
                                            outp, Dn, 0, bo, (int)Dn, 1.0f, 1);
}
